// CSWinB_90520730730921
// MI455X (gfx1250) — matrix operand order repaired and checked by reading, compiled, not run on hardware
//
#include <hip/hip_runtime.h>
#include <hip/hip_bf16.h>
#include <stdint.h>

// ---------------------------------------------------------------------------
// CSWin block for MI455X (gfx1250, wave32, WMMA bf16 + async-to-LDS path)
// ---------------------------------------------------------------------------

typedef __attribute__((ext_vector_type(16))) __bf16 v16bf;
typedef __attribute__((ext_vector_type(8)))  float  v8f;

#define BQ 4
#define LL 4096          // H*W
#define MROWS 16384      // B*L
#define C2 512
#define QKVC 1536
// 1/sqrt(32) * log2(e): scores arrive pre-scaled for exp2
#define ATT_SCALE_LOG2E 0.25500444058297157f

static __device__ __forceinline__ unsigned short f2bf(float f) {
  union { __bf16 b; unsigned short u; } c; c.b = (__bf16)f;
  return c.u;
}
static __device__ __forceinline__ float bf2f(unsigned short h) {
  union { unsigned u; float f; } c; c.u = ((unsigned)h) << 16;
  return c.f;
}
static __device__ __forceinline__ __bf16 bfraw(unsigned short h) {
  union { unsigned short u; __bf16 b; } c; c.u = h;
  return c.b;
}
static __device__ __forceinline__ v8f zero8() {
  v8f z = {0.f,0.f,0.f,0.f,0.f,0.f,0.f,0.f};
  return z;
}
// low 32 bits of a generic LDS pointer == LDS byte offset (ISA 10.2)
static __device__ __forceinline__ unsigned lds_off(const void* p) {
  return (unsigned)(size_t)p;
}
// async 16B copy global -> LDS (per-lane), tracked by ASYNCcnt
static __device__ __forceinline__ void async_ld_b128(unsigned ldsdst, const void* gsrc) {
  asm volatile("global_load_async_to_lds_b128 %0, %1, off"
               :: "v"(ldsdst), "v"((unsigned long long)(size_t)gsrc)
               : "memory");
}
static __device__ __forceinline__ void wait_async0() {
  asm volatile("s_wait_asynccnt 0x0" ::: "memory");
}

// ---------------------------------------------------------------------------
// bf16 WMMA GEMM: C[M,N] = A[M,K](lda) @ B[K,N]  (+bias +residual)
// Block = 128 threads (4 waves). Block tile 128(M) x 64(N), K-step 32.
// Each wave: 32 rows x 4 WMMA n-tiles (8 WMMAs per K-step).
// A tile staged with GLOBAL_LOAD_ASYNC_TO_LDS_B128; B tile sync (transposed).
// ---------------------------------------------------------------------------
template<bool HAS_BIAS, bool HAS_RES, bool OUTF_EN, bool OUTB_EN, bool TRANS>
__global__ __launch_bounds__(128) void gemm_bf16_kernel(
    const unsigned short* __restrict__ A, int lda,
    const unsigned short* __restrict__ Bw,   // [K][N] row-major
    int M, int N, int K,
    const float* __restrict__ bias,
    const float* __restrict__ residual,      // [M][N] f32
    float* __restrict__ outF,
    unsigned short* __restrict__ outB,
    int trL)                                 // TRANS: outF[((row/trL)*N+col)*trL + row%trL]
{
  __shared__ unsigned short As[128 * 40];    // [row][k], stride 40 bf16 (80B rows)
  __shared__ unsigned short Bs[64 * 40];     // [n][k] transposed, stride 40

  const int tid  = threadIdx.x;
  const int wave = tid >> 5;
  const int lane = tid & 31;
  const int m0 = blockIdx.y * 128;
  const int n0 = blockIdx.x * 64;

  v8f acc[2][4];
  #pragma unroll
  for (int mi = 0; mi < 2; mi++)
    #pragma unroll
    for (int nt = 0; nt < 4; nt++) acc[mi][nt] = zero8();

  for (int k0 = 0; k0 < K; k0 += 32) {
    __syncthreads();
    { // A tile via async copy: 128 rows x 32 cols = 512 x 16B chunks
      #pragma unroll
      for (int i = 0; i < 4; i++) {
        int chunk = tid + 128 * i;
        int row = chunk >> 2;
        int cc  = (chunk & 3) * 8;           // 8 bf16 = 16B per chunk
        async_ld_b128(lds_off(As + row * 40 + cc),
                      A + (size_t)(m0 + row) * lda + k0 + cc);
      }
    }
    { // B tile transposed (sync): Bs[n][k] <- Bw[k0+k][n0+n]
      int kk = tid >> 2;
      int nc = (tid & 3) * 16;
      const uint4* g = (const uint4*)(Bw + (size_t)(k0 + kk) * N + n0 + nc);
      uint4 b0 = g[0], b1 = g[1];
      unsigned short tmp[16] __attribute__((aligned(16)));
      *(uint4*)&tmp[0] = b0; *(uint4*)&tmp[8] = b1;
      #pragma unroll
      for (int i = 0; i < 16; i++) Bs[(nc + i) * 40 + kk] = tmp[i];
    }
    wait_async0();
    __syncthreads();

    // A fragments: rows m0 + 32*wave + {0..15, 16..31}
    v16bf af[2];
    #pragma unroll
    for (int mi = 0; mi < 2; mi++) {
      const unsigned short* ap = As + (32 * wave + 16 * mi + (lane & 15)) * 40
                                 + ((lane < 16) ? 0 : 8);
      #pragma unroll
      for (int i = 0; i < 8; i++) {
        af[mi][i]     = bfraw(ap[i]);        // K = khalf + i
        af[mi][8 + i] = bfraw(ap[16 + i]);   // K = 16 + khalf + i
      }
    }
    #pragma unroll
    for (int nt = 0; nt < 4; nt++) {
      v16bf bf;
      const unsigned short* bp = Bs + (nt * 16 + (lane & 15)) * 40 + ((lane < 16) ? 0 : 8);
      #pragma unroll
      for (int i = 0; i < 16; i++) bf[i] = bfraw(bp[i < 8 ? i : 8 + i]);   // K = khalf + i, 16 + khalf + (i-8)
      acc[0][nt] = __builtin_amdgcn_wmma_f32_16x16x32_bf16(
          false, af[0], false, bf, (short)0, acc[0][nt], false, false);
      acc[1][nt] = __builtin_amdgcn_wmma_f32_16x16x32_bf16(
          false, af[1], false, bf, (short)0, acc[1][nt], false, false);
    }
  }

  // Epilogue (C/D layout: VGPR r, lanes 0-15 -> M=r; lanes 16-31 -> M=8+r)
  int cbase = n0 + (lane & 15);
  #pragma unroll
  for (int mi = 0; mi < 2; mi++) {
    int rbase = m0 + 32 * wave + 16 * mi + ((lane >> 4) << 3);
    #pragma unroll
    for (int nt = 0; nt < 4; nt++) {
      int col = cbase + nt * 16;
      float bv = HAS_BIAS ? bias[col] : 0.0f;
      #pragma unroll
      for (int r = 0; r < 8; r++) {
        int row = rbase + r;
        float val = acc[mi][nt][r] + bv;
        if (HAS_RES) val += residual[(size_t)row * N + col];
        if (TRANS) {
          int bb = row / trL, l = row - bb * trL;
          outF[((size_t)bb * N + col) * trL + l] = val;
        } else {
          if (OUTF_EN) outF[(size_t)row * N + col] = val;
          if (OUTB_EN) outB[(size_t)row * N + col] = f2bf(val);
        }
      }
    }
  }
}

// ---------------------------------------------------------------------------
// Elementwise helpers
// ---------------------------------------------------------------------------
__global__ void cvt_f32_bf16(const float* __restrict__ s,
                             unsigned short* __restrict__ d, int n) {
  int i = blockIdx.x * 256 + threadIdx.x;
  if (i < n) d[i] = f2bf(s[i]);
}

// x[B,256,L] -> xt_bf16[B*L,256]
__global__ void xpose_x(const float* __restrict__ x, unsigned short* __restrict__ xt) {
  int o = blockIdx.x * 256 + threadIdx.x;       // (b*L+l)*256 + c
  int c = o & 255;
  int l = (o >> 8) & 4095;
  int b = o >> 20;
  xt[o] = f2bf(x[((size_t)(b * 256 + c)) * LL + l]);
}

// LayerNorm over 512 channels, one wave per row (16 elems/lane)
__global__ __launch_bounds__(256) void ln_kernel(const float* __restrict__ x2,
                                                 const float* __restrict__ g,
                                                 const float* __restrict__ bsh,
                                                 unsigned short* __restrict__ img) {
  int wave = threadIdx.x >> 5, lane = threadIdx.x & 31;
  int row = blockIdx.x * 8 + wave;
  const float* xr = x2 + (size_t)row * C2;
  float v[16], s = 0.f, sq = 0.f;
  #pragma unroll
  for (int i = 0; i < 16; i++) {
    v[i] = xr[lane + 32 * i];
    s += v[i]; sq += v[i] * v[i];
  }
  #pragma unroll
  for (int m = 1; m < 32; m <<= 1) {
    s  += __shfl_xor(s, m, 32);
    sq += __shfl_xor(sq, m, 32);
  }
  float mu   = s * (1.f / 512.f);
  float var  = sq * (1.f / 512.f) - mu * mu;
  float rstd = rsqrtf(var + 1e-5f);
  unsigned short* orow = img + (size_t)row * C2;
  #pragma unroll
  for (int i = 0; i < 16; i++) {
    int c = lane + 32 * i;
    orow[c] = f2bf((v[i] - mu) * rstd * g[c] + bsh[c]);
  }
}

// Depthwise 3x3 LePE per window; pre-fills o32 = lepe + bias
__global__ void lepe_kernel(const unsigned short* __restrict__ qkv,
                            const float* __restrict__ cw0, const float* __restrict__ cb0,
                            const float* __restrict__ cw1, const float* __restrict__ cb1,
                            float* __restrict__ o) {
  int idx = blockIdx.x * 256 + threadIdx.x;     // (b*L+l)*512 + c
  int c  = idx & 511;
  int l  = (idx >> 9) & 4095;
  int b  = idx >> 21;
  int br = c >> 8, ch = c & 255;
  const float* cw = br ? cw1 : cw0;
  const float* cb = br ? cb1 : cb0;
  int y = l >> 6, x = l & 63;
  int Hs, Ws, i, j, wy, wx;
  if (br == 0) { Hs = 64; Ws = 8;  i = y;     j = x & 7; wy = 0;      wx = x & ~7; }
  else         { Hs = 8;  Ws = 64; i = y & 7; j = x;     wy = y & ~7; wx = 0;      }
  float acc = cb[ch];
  #pragma unroll
  for (int di = 0; di < 3; di++) {
    int ii = i + di - 1;
    if (ii < 0 || ii >= Hs) continue;
    #pragma unroll
    for (int dj = 0; dj < 3; dj++) {
      int jj = j + dj - 1;
      if (jj < 0 || jj >= Ws) continue;
      int ll = (wy + ii) * 64 + (wx + jj);
      float vv = bf2f(qkv[((size_t)(b * LL + ll)) * QKVC + 1024 + c]);
      acc += vv * cw[ch * 9 + di * 3 + dj];
    }
  }
  o[idx] = acc;
}

// ---------------------------------------------------------------------------
// Attention per (branch, window, head): Q,K,V = [512,32] bf16.
// Transposed-score trick: S^T = K_tile x Q^T puts each lane's D-layout
// elements exactly where the A-layout P fragment needs them -> P = exp2(S^T)
// is a pure register transform (no LDS scratch for P).  log2(e) folded into
// the Q pre-scale so exp is a bare v_exp_f32.  Row sums via osum += P x Ones
// on the matrix engine.  5 WMMAs per 32-col step.
// Block = 256 threads (8 waves), each wave = 16 Q rows; grid.x=4 covers 512.
// ---------------------------------------------------------------------------
static __device__ __forceinline__ int tok2l(int t, int br, int w) {
  if (br == 0) return ((t >> 3) << 6) + (w << 3) + (t & 7);      // H_sp=64, W_sp=8
  return (((w << 3) + (t >> 6)) << 6) + (t & 63);               // H_sp=8,  W_sp=64
}

__global__ __launch_bounds__(256) void attn_kernel(const unsigned short* __restrict__ qkv,
                                                   float* __restrict__ o32) {
  __shared__ unsigned short Ks[256 * 40];       // [tok][ch]  stride 40 (80B rows)
  __shared__ unsigned short Vt[32 * 264];       // [ch][tok]  stride 264

  const int tid = threadIdx.x, wave = tid >> 5, lane = tid & 31;
  const int br = blockIdx.z;
  const int h  = blockIdx.y & 7;
  const int wg = blockIdx.y >> 3;
  const int b  = wg >> 3, w = wg & 7;
  const int chq = br * 256 + h * 32;

  // Q fragment for this wave's 16 rows, as B-matrix (32ch x 16q): lane holds
  // col q = lane%16, K = khalf8 + i and 16 + khalf8 + (i-8).  Pre-scaled by log2e/sqrt(hd).
  const int qt0 = blockIdx.x * 128 + wave * 16;
  v16bf qf;
  {
    int t = qt0 + (lane & 15);
    int l = tok2l(t, br, w);
    const unsigned short* qp = qkv + ((size_t)(b * LL + l)) * QKVC + chq
                               + ((lane < 16) ? 0 : 8);
    #pragma unroll
    for (int i = 0; i < 16; i++)
      qf[i] = (__bf16)(bf2f(qp[i < 8 ? i : 8 + i]) * ATT_SCALE_LOG2E);
  }
  // all-ones B fragment (for row sums of P via WMMA)
  v16bf onesf;
  #pragma unroll
  for (int v = 0; v < 16; v++) onesf[v] = bfraw((unsigned short)0x3F80);

  v8f o0 = zero8(), o1 = zero8(), osum = zero8();

  for (int chunk = 0; chunk < 2; chunk++) {
    __syncthreads();
    { // stage 256 tokens of K (async) and V (sync transpose), 1 token/thread
      int lt = tid;
      int t  = chunk * 256 + lt;
      int l  = tok2l(t, br, w);
      const unsigned short* kg = qkv + ((size_t)(b * LL + l)) * QKVC + 512 + chq;
      const unsigned short* vg = qkv + ((size_t)(b * LL + l)) * QKVC + 1024 + chq;
      unsigned kdst = lds_off(Ks + lt * 40);
      #pragma unroll
      for (int i = 0; i < 4; i++)
        async_ld_b128(kdst + 16 * i, kg + 8 * i);
      uint4 v0 = ((const uint4*)vg)[0], v1 = ((const uint4*)vg)[1];
      uint4 v2 = ((const uint4*)vg)[2], v3 = ((const uint4*)vg)[3];
      unsigned short va[32] __attribute__((aligned(16)));
      *(uint4*)&va[0]  = v0; *(uint4*)&va[8]  = v1;
      *(uint4*)&va[16] = v2; *(uint4*)&va[24] = v3;
      #pragma unroll
      for (int c = 0; c < 32; c++) Vt[c * 264 + lt] = va[c];
    }
    wait_async0();
    __syncthreads();

    for (int j0 = 0; j0 < 256; j0 += 32) {
      // K tiles as A-matrix (16 tok x 32 ch): lane holds row tok = j0+lane%16,
      // K = khalf8 + i and 16 + khalf8 + i (two contiguous 16B chunks)
      v16bf ka0, ka1;
      {
        const unsigned short* kp = Ks + (j0 + (lane & 15)) * 40 + ((lane < 16) ? 0 : 8);
        const unsigned short* kq = kp + 16 * 40;
        #pragma unroll
        for (int i = 0; i < 8; i++) {
          ka0[i] = bfraw(kp[i]); ka0[8 + i] = bfraw(kp[16 + i]);
          ka1[i] = bfraw(kq[i]); ka1[8 + i] = bfraw(kq[16 + i]);
        }
      }
      // S^T tiles: D[m=j][n=q], already scaled for exp2
      v8f s0t = __builtin_amdgcn_wmma_f32_16x16x32_bf16(false, ka0, false, qf, (short)0, zero8(), false, false);
      v8f s1t = __builtin_amdgcn_wmma_f32_16x16x32_bf16(false, ka1, false, qf, (short)0, zero8(), false, false);

      // P = exp2(S') in A-layout directly from registers:
      // pf[i] <- frag0 r=i, pf[8+i] <- frag1 r=i  (per-lane j-sets align)
      v16bf pf;
      #pragma unroll
      for (int i = 0; i < 8; i++) {
        pf[i]     = (__bf16)__builtin_amdgcn_exp2f(s0t[i]);
        pf[8 + i] = (__bf16)__builtin_amdgcn_exp2f(s1t[i]);
      }
      // V fragments: B[k][d] = Vt[d][j0+k]
      v16bf vf0, vf1;
      {
        const unsigned short* vp = Vt + (lane & 15) * 264 + j0 + ((lane < 16) ? 0 : 8);
        #pragma unroll
        for (int i = 0; i < 16; i++) {
          vf0[i] = bfraw(vp[i < 8 ? i : 8 + i]);
          vf1[i] = bfraw(vp[16 * 264 + (i < 8 ? i : 8 + i)]);
        }
      }
      o0   = __builtin_amdgcn_wmma_f32_16x16x32_bf16(false, pf, false, vf0,   (short)0, o0,   false, false);
      o1   = __builtin_amdgcn_wmma_f32_16x16x32_bf16(false, pf, false, vf1,   (short)0, o1,   false, false);
      osum = __builtin_amdgcn_wmma_f32_16x16x32_bf16(false, pf, false, onesf, (short)0, osum, false, false);
    }
  }

  // finalize: divide by row sums (osum columns all hold the row sum, already
  // in matching C/D layout), add onto pre-filled LePE values
  #pragma unroll
  for (int r = 0; r < 8; r++) {
    float inv = 1.f / osum[r];
    int t = qt0 + r + ((lane >> 4) << 3);
    int l = tok2l(t, br, w);
    size_t base = ((size_t)(b * LL + l)) * C2 + chq;
    int d = lane & 15;
    o32[base + d]      += o0[r] * inv;
    o32[base + 16 + d] += o1[r] * inv;
  }
}

// ---------------------------------------------------------------------------
// Launch
// ---------------------------------------------------------------------------
extern "C" void kernel_launch(void* const* d_in, const int* in_sizes, int n_in,
                              void* d_out, int out_size, void* d_ws, size_t ws_size,
                              hipStream_t stream) {
  const float* x       = (const float*)d_in[0];
  const float* w_embed = (const float*)d_in[1];
  const float* g1      = (const float*)d_in[2];
  const float* b1      = (const float*)d_in[3];
  const float* w_qkv   = (const float*)d_in[4];
  const float* cw0     = (const float*)d_in[5];
  const float* cb0     = (const float*)d_in[6];
  const float* cw1     = (const float*)d_in[7];
  const float* cb1     = (const float*)d_in[8];
  const float* w_proj  = (const float*)d_in[9];
  const float* b_proj  = (const float*)d_in[10];
  const float* w_out   = (const float*)d_in[11];
  float* out = (float*)d_out;

  char* base = (char*)d_ws;
  size_t off = 0;
  auto alloc = [&](size_t bytes) -> void* {
    void* p = base + off;
    off = (off + bytes + 255) & ~(size_t)255;
    return p;
  };
  unsigned short* xt   = (unsigned short*)alloc((size_t)MROWS * 256 * 2);
  float*          x2   = (float*)         alloc((size_t)MROWS * 512 * 4);
  unsigned short* img  = (unsigned short*)alloc((size_t)MROWS * 512 * 2);
  unsigned short* qkvb = (unsigned short*)alloc((size_t)MROWS * 1536 * 2);
  float*          o32  = (float*)         alloc((size_t)MROWS * 512 * 4);
  unsigned short* ob   = (unsigned short*)alloc((size_t)MROWS * 512 * 2);
  unsigned short* xob  = (unsigned short*)alloc((size_t)MROWS * 512 * 2);
  unsigned short* web  = (unsigned short*)alloc((size_t)256 * 512 * 2);
  unsigned short* wqb  = (unsigned short*)alloc((size_t)512 * 1536 * 2);
  unsigned short* wpb  = (unsigned short*)alloc((size_t)512 * 512 * 2);
  unsigned short* wob  = (unsigned short*)alloc((size_t)512 * 256 * 2);

  // weight + input conversions to bf16
  cvt_f32_bf16<<<(256 * 512 + 255) / 256, 256, 0, stream>>>(w_embed, web, 256 * 512);
  cvt_f32_bf16<<<(512 * 1536 + 255) / 256, 256, 0, stream>>>(w_qkv, wqb, 512 * 1536);
  cvt_f32_bf16<<<(512 * 512 + 255) / 256, 256, 0, stream>>>(w_proj, wpb, 512 * 512);
  cvt_f32_bf16<<<(512 * 256 + 255) / 256, 256, 0, stream>>>(w_out, wob, 512 * 256);
  xpose_x<<<(MROWS * 256) / 256, 256, 0, stream>>>(x, xt);

  // x2 = xt @ w_embed (f32 out for residual)
  gemm_bf16_kernel<false,false,true,false,false>
      <<<dim3(512 / 64, MROWS / 128), 128, 0, stream>>>(
      xt, 256, web, MROWS, 512, 256, nullptr, nullptr, x2, nullptr, 0);
  // img = LN(x2) in bf16
  ln_kernel<<<MROWS / 8, 256, 0, stream>>>(x2, g1, b1, img);
  // qkv = img @ w_qkv (bf16 out)
  gemm_bf16_kernel<false,false,false,true,false>
      <<<dim3(1536 / 64, MROWS / 128), 128, 0, stream>>>(
      img, 512, wqb, MROWS, 1536, 512, nullptr, nullptr, nullptr, qkvb, 0);
  // o32 = LePE(v) (both branches), then attention adds into it
  lepe_kernel<<<(MROWS * 512) / 256, 256, 0, stream>>>(qkvb, cw0, cb0, cw1, cb1, o32);
  attn_kernel<<<dim3(4, 256, 2), 256, 0, stream>>>(qkvb, o32);
  cvt_f32_bf16<<<(MROWS * 512 + 255) / 256, 256, 0, stream>>>(o32, ob, MROWS * 512);
  // xo = x2 + o32 @ w_proj + b_proj (bf16 out)
  gemm_bf16_kernel<true,true,false,true,false>
      <<<dim3(512 / 64, MROWS / 128), 128, 0, stream>>>(
      ob, 512, wpb, MROWS, 512, 512, b_proj, x2, nullptr, xob, 0);
  // out = (xo @ w_out)^T -> [B,256,H,W]
  gemm_bf16_kernel<false,false,true,false,true>
      <<<dim3(256 / 64, MROWS / 128), 128, 0, stream>>>(
      xob, 512, wob, MROWS, 256, 512, nullptr, nullptr, out, nullptr, LL);
  // vout = (v @ w_out)^T -> [B,256,H,W]
  gemm_bf16_kernel<false,false,true,false,true>
      <<<dim3(256 / 64, MROWS / 128), 128, 0, stream>>>(
      qkvb + 1024, 1536, wob, MROWS, 256, 512, nullptr, nullptr,
      out + (size_t)BQ * 256 * LL, nullptr, LL);
}
